// GraphAttentionBlock_49503793054481
// MI455X (gfx1250) — matrix operand order repaired and checked by reading, not compiled, not run on hardware
//
#include <hip/hip_runtime.h>
#include <hip/hip_bf16.h>
#include <math.h>

// ---------------------------------------------------------------------------
// GraphAttentionBlock for MI455X (gfx1250).
// B=4, N=256, DX=DY=128, DE=64, H=8, HD=16.
// Heavy edge GEMMs run on v_wmma_f32_16x16x32_bf16 (f32 accumulate); the whole
// per-(b,i) edge pipeline (ea linears -> logits -> softmax/attn -> eo / no
// linears) is fused in one workgroup with the 256x128 logit tile in LDS.
// ---------------------------------------------------------------------------

#define BB 4
#define NN 256
#define DX 128
#define DE 64
#define DY 128

typedef float v8f  __attribute__((ext_vector_type(8)));
typedef __bf16 v16bf __attribute__((ext_vector_type(16)));

union BFrag {
    v16bf v;
    unsigned short s[16];
};

__device__ __forceinline__ unsigned short f2bf(float f) {
    unsigned u = __float_as_uint(f);
    unsigned r = u + 0x7FFFu + ((u >> 16) & 1u);   // round-to-nearest-even
    return (unsigned short)(r >> 16);
}

#define V8F_ZERO {0.f,0.f,0.f,0.f,0.f,0.f,0.f,0.f}

// ---------------------------------------------------------------------------
// K0: f32 -> bf16 weight conversion (run once per weight matrix)
// ---------------------------------------------------------------------------
__global__ void cvt_bf16_kernel(const float* __restrict__ src,
                                unsigned short* __restrict__ dst, int n) {
    int i = blockIdx.x * 256 + threadIdx.x;
    if (i < n) dst[i] = f2bf(src[i]);
}

// ---------------------------------------------------------------------------
// K1: y-conditioned affine parameters: ge_add(y), ge_mul(y), gn_add(y), gn_mul(y)
// grid = B, block = 128
// ---------------------------------------------------------------------------
__global__ __launch_bounds__(128)
void y_affine_kernel(const float* __restrict__ y,
                     const float* geaW, const float* geab,
                     const float* gemW, const float* gemb,
                     const float* gnaW, const float* gnab,
                     const float* gnmW, const float* gnmb,
                     float* geA, float* geM, float* gnA, float* gnM) {
    __shared__ float sy[DY];
    int b = blockIdx.x, t = threadIdx.x;
    sy[t] = y[b * DY + t];
    __syncthreads();
    float a0 = geab[t], a1 = gemb[t], a2 = gnab[t], a3 = gnmb[t];
    #pragma unroll 4
    for (int k = 0; k < DY; ++k) {
        float yv = sy[k];
        a0 += geaW[t * DY + k] * yv;
        a1 += gemW[t * DY + k] * yv;
        a2 += gnaW[t * DY + k] * yv;
        a3 += gnmW[t * DY + k] * yv;
    }
    geA[b * DX + t] = a0;
    geM[b * DX + t] = a1;
    gnA[b * DX + t] = a2;
    gnM[b * DX + t] = a3;
}

// ---------------------------------------------------------------------------
// K2: Q/K/V linear via WMMA bf16: out[r,c] = (X[r,:]@W[c,:] + b[c]) * nm[r]
// M = B*N = 1024 rows, Nout = 128, K = 128. grid = 64 m-tiles, block = 256
// (8 waves; wave w owns column tile w).
// ---------------------------------------------------------------------------
__global__ __launch_bounds__(256)
void qkv_wmma_kernel(const float* __restrict__ X,
                     const unsigned short* __restrict__ Wbf,
                     const float* __restrict__ bias,
                     const float* __restrict__ nmask,
                     float* __restrict__ out) {
    int mt   = blockIdx.x;
    int wave = threadIdx.x >> 5;
    int lane = threadIdx.x & 31;
    int lh   = lane >> 4;
    int ln   = lane & 15;

    v8f acc = V8F_ZERO;
    int arow = mt * 16 + ln;
    #pragma unroll
    for (int ks = 0; ks < 128; ks += 32) {
        BFrag a;
        const float* p0 = X + (size_t)arow * DX + ks + (lh ? 8 : 0);
        #pragma unroll
        for (int e = 0; e < 8; ++e) {
            a.s[e]     = f2bf(p0[e]);
            a.s[8 + e] = f2bf(p0[16 + e]);
        }
        BFrag bw;
        const unsigned short* wp = Wbf + (size_t)(wave * 16 + ln) * DX + ks + (lh ? 8 : 0);
        #pragma unroll
        for (int e = 0; e < 16; ++e) bw.s[e] = wp[e < 8 ? e : e + 8];
        acc = __builtin_amdgcn_wmma_f32_16x16x32_bf16(false, a.v, false, bw.v,
                                                      (short)0, acc, false, false);
    }
    #pragma unroll
    for (int v = 0; v < 8; ++v) {
        int r = mt * 16 + v + lh * 8;
        int c = wave * 16 + ln;
        out[(size_t)r * DX + c] = (acc[v] + bias[c]) * nmask[r];
    }
}

// ---------------------------------------------------------------------------
// K3: fused edge pipeline. grid = B*N (one workgroup per (b,i)), block = 256.
// Stage 1: logits[j,c] via WMMA (K=64)  -> LDS
// Stage 2: updated_E = (geA+(geM+1)*logits) @ eo^T via WMMA (K=128)
// Stage 3: per-channel softmax over j, attn@V, gn affine, no_W linear
// ---------------------------------------------------------------------------
__global__ __launch_bounds__(256)
void edge_attn_kernel(const float* __restrict__ E,
                      const float* __restrict__ Q,
                      const float* __restrict__ Kmat,
                      const float* __restrict__ V,
                      const float* __restrict__ nmask,
                      const unsigned short* __restrict__ eamul_bf,
                      const unsigned short* __restrict__ eaadd_bf,
                      const unsigned short* __restrict__ eo_bf,
                      const float* __restrict__ ea_mul_b,
                      const float* __restrict__ ea_add_b,
                      const float* __restrict__ eo_b,
                      const float* __restrict__ geA,
                      const float* __restrict__ geM,
                      const float* __restrict__ gnA,
                      const float* __restrict__ gnM,
                      const float* __restrict__ no_W,
                      const float* __restrict__ no_b,
                      float* __restrict__ outX,
                      float* __restrict__ outE) {
    __shared__ float s_logits[NN * DX];   // 128 KB
    __shared__ float s_nm[NN];
    __shared__ float s_qi[DX];
    __shared__ float s_geA[DX];
    __shared__ float s_geM[DX];
    __shared__ float s_uX[DX];

    int b = blockIdx.x >> 8;
    int i = blockIdx.x & 255;
    int tid  = threadIdx.x;
    int wave = tid >> 5;
    int lane = tid & 31;
    int lh   = lane >> 4;
    int ln   = lane & 15;

    if (tid < DX) {
        s_qi[tid]  = Q[((size_t)(b * NN + i)) * DX + tid];
        s_geA[tid] = geA[b * DX + tid];
        s_geM[tid] = geM[b * DX + tid];
    }
    s_nm[tid] = nmask[b * NN + tid];
    __syncthreads();

    const float* Eb = E + ((size_t)(b * NN + i)) * NN * DE;

    // ---- Stage 1: logits -> LDS -------------------------------------------
    for (int t = wave; t < 16 * 8; t += 8) {          // 16 j-tiles x 8 c-tiles
        int jt = t >> 3, ct = t & 7;
        v8f am = V8F_ZERO, aa = V8F_ZERO;
        #pragma unroll
        for (int ks = 0; ks < DE; ks += 32) {
            BFrag a;
            const float* p0 = Eb + (size_t)(jt * 16 + ln) * DE + ks + (lh ? 8 : 0);
            #pragma unroll
            for (int e = 0; e < 8; ++e) {
                a.s[e]     = f2bf(p0[e]);
                a.s[8 + e] = f2bf(p0[16 + e]);
            }
            int n  = ct * 16 + ln;
            int kk = ks + (lh ? 8 : 0);
            BFrag bm, ba;
            const unsigned short* pm = eamul_bf + (size_t)n * DE + kk;
            const unsigned short* pa = eaadd_bf + (size_t)n * DE + kk;
            #pragma unroll
            for (int e = 0; e < 16; ++e) { bm.s[e] = pm[e < 8 ? e : e + 8]; ba.s[e] = pa[e < 8 ? e : e + 8]; }
            am = __builtin_amdgcn_wmma_f32_16x16x32_bf16(false, a.v, false, bm.v,
                                                         (short)0, am, false, false);
            aa = __builtin_amdgcn_wmma_f32_16x16x32_bf16(false, a.v, false, ba.v,
                                                         (short)0, aa, false, false);
        }
        int c = ct * 16 + ln;
        #pragma unroll
        for (int v = 0; v < 8; ++v) {
            int j = jt * 16 + v + lh * 8;
            float mul = am[v] + ea_mul_b[c];
            float add = aa[v] + ea_add_b[c];
            float raw = s_qi[c] * Kmat[((size_t)(b * NN + j)) * DX + c] * 0.25f; // 1/sqrt(16)
            s_logits[j * DX + c] = raw * (mul + 1.f) + add;
        }
    }
    __syncthreads();

    float nm_i = s_nm[i];

    // ---- Stage 2: updated_E ------------------------------------------------
    for (int t = wave; t < 16 * 4; t += 8) {          // 16 j-tiles x 4 e-tiles
        int jt = t >> 2, et = t & 3;
        v8f acc = V8F_ZERO;
        #pragma unroll
        for (int ks = 0; ks < DX; ks += 32) {
            BFrag a;
            int j = jt * 16 + ln;
            int kb = ks + (lh ? 8 : 0);
            #pragma unroll
            for (int e = 0; e < 8; ++e) {
                int c0 = kb + e, c1 = kb + 16 + e;
                a.s[e]     = f2bf(s_geA[c0] + (s_geM[c0] + 1.f) * s_logits[j * DX + c0]);
                a.s[8 + e] = f2bf(s_geA[c1] + (s_geM[c1] + 1.f) * s_logits[j * DX + c1]);
            }
            BFrag bw;
            const unsigned short* pw = eo_bf + (size_t)(et * 16 + ln) * DX + ks + (lh ? 8 : 0);
            #pragma unroll
            for (int e = 0; e < 16; ++e) bw.s[e] = pw[e < 8 ? e : e + 8];
            acc = __builtin_amdgcn_wmma_f32_16x16x32_bf16(false, a.v, false, bw.v,
                                                          (short)0, acc, false, false);
        }
        int ec = et * 16 + ln;
        #pragma unroll
        for (int v = 0; v < 8; ++v) {
            int j = jt * 16 + v + lh * 8;
            outE[(((size_t)(b * NN + i)) * NN + j) * DE + ec] =
                (acc[v] + eo_b[ec]) * nm_i * s_nm[j];
        }
    }

    // ---- Stage 3: softmax over j, attn@V, gn affine -----------------------
    if (tid < DX) {
        int c = tid;
        float m = -3.402823466e38f;
        for (int j = 0; j < NN; ++j)
            if (s_nm[j] != 0.f) m = fmaxf(m, s_logits[j * DX + c]);
        float ssum = 0.f, acc = 0.f;
        const float* Vb = V + ((size_t)b * NN) * DX + c;
        for (int j = 0; j < NN; ++j) {
            if (s_nm[j] != 0.f) {
                float e = __expf(s_logits[j * DX + c] - m);
                ssum += e;
                acc  += e * Vb[(size_t)j * DX];
            }
        }
        float wv = acc / ssum;
        s_uX[c] = gnA[b * DX + c] + (gnM[b * DX + c] + 1.f) * wv;
    }
    __syncthreads();
    if (tid < DX) {
        float sum = no_b[tid];
        const float* wr = no_W + (size_t)tid * DX;
        #pragma unroll 4
        for (int c = 0; c < DX; ++c) sum += wr[c] * s_uX[c];
        outX[((size_t)(b * NN + i)) * DX + tid] = sum * nm_i;
    }
}

// ---------------------------------------------------------------------------
// K4: node statistics over N. grid = B, block = 128 (one thread per channel)
// node_stats layout per b: [mean(128) | min(128) | max(128) | std(128)]
// ---------------------------------------------------------------------------
__global__ __launch_bounds__(128)
void xstats_kernel(const float* __restrict__ X, float* __restrict__ node_stats) {
    int b = blockIdx.x, c = threadIdx.x;
    float s = 0.f, sq = 0.f, mn = 3.402823466e38f, mx = -3.402823466e38f;
    for (int n = 0; n < NN; ++n) {
        float v = X[((size_t)(b * NN + n)) * DX + c];
        s += v; sq += v * v;
        mn = fminf(mn, v); mx = fmaxf(mx, v);
    }
    float mean = s / (float)NN;
    float var  = (sq - s * s / (float)NN) / (float)(NN - 1);
    float* o = node_stats + b * 4 * DX;
    o[c]            = mean;
    o[DX + c]       = mn;
    o[2 * DX + c]   = mx;
    o[3 * DX + c]   = sqrtf(fmaxf(var, 0.f));
}

// ---------------------------------------------------------------------------
// K5: edge statistics over N*N. grid = B*DE, block = 256, LDS tree reduce.
// edge_stats layout per b: [mean(64) | min(64) | max(64) | std(64)]
// ---------------------------------------------------------------------------
__global__ __launch_bounds__(256)
void estats_kernel(const float* __restrict__ E, float* __restrict__ edge_stats) {
    __shared__ float ls[256], lq[256], lmn[256], lmx[256];
    int b = blockIdx.x >> 6;
    int c = blockIdx.x & 63;
    int t = threadIdx.x;
    const int M = NN * NN;
    float s = 0.f, sq = 0.f, mn = 3.402823466e38f, mx = -3.402823466e38f;
    for (int ij = t; ij < M; ij += 256) {
        float v = E[((size_t)b * M + ij) * DE + c];
        s += v; sq += v * v;
        mn = fminf(mn, v); mx = fmaxf(mx, v);
    }
    ls[t] = s; lq[t] = sq; lmn[t] = mn; lmx[t] = mx;
    __syncthreads();
    for (int off = 128; off > 0; off >>= 1) {
        if (t < off) {
            ls[t] += ls[t + off];
            lq[t] += lq[t + off];
            lmn[t] = fminf(lmn[t], lmn[t + off]);
            lmx[t] = fmaxf(lmx[t], lmx[t + off]);
        }
        __syncthreads();
    }
    if (t == 0) {
        float n = (float)M;
        float mean = ls[0] / n;
        float var  = (lq[0] - ls[0] * ls[0] / n) / (n - 1.f);
        float* o = edge_stats + b * 4 * DE;
        o[c]            = mean;
        o[DE + c]       = lmn[0];
        o[2 * DE + c]   = lmx[0];
        o[3 * DE + c]   = sqrtf(fmaxf(var, 0.f));
    }
}

// ---------------------------------------------------------------------------
// K6: head MLP: uy = go2(relu(go1(gs(y) + na(node_stats) + eag(edge_stats))))
// grid = B, block = 128
// ---------------------------------------------------------------------------
__global__ __launch_bounds__(128)
void head_mlp_kernel(const float* __restrict__ y,
                     const float* __restrict__ node_stats,
                     const float* __restrict__ edge_stats,
                     const float* gsW, const float* gsb,
                     const float* naW, const float* nab,
                     const float* eagW, const float* eagb,
                     const float* go1W, const float* go1b,
                     const float* go2W, const float* go2b,
                     float* __restrict__ out_uy) {
    __shared__ float s0[DY], s1[DY];
    int b = blockIdx.x, t = threadIdx.x;
    float v = gsb[t] + nab[t] + eagb[t];
    for (int k = 0; k < DY; ++k)      v += gsW[t * DY + k]      * y[b * DY + k];
    for (int k = 0; k < 4 * DX; ++k)  v += naW[t * 4 * DX + k]  * node_stats[b * 4 * DX + k];
    for (int k = 0; k < 4 * DE; ++k)  v += eagW[t * 4 * DE + k] * edge_stats[b * 4 * DE + k];
    s0[t] = v;
    __syncthreads();
    float h = go1b[t];
    for (int k = 0; k < DY; ++k) h += go1W[t * DY + k] * s0[k];
    s1[t] = fmaxf(h, 0.f);
    __syncthreads();
    float o = go2b[t];
    for (int k = 0; k < DY; ++k) o += go2W[t * DY + k] * s1[k];
    out_uy[b * DY + t] = o;
}

// ---------------------------------------------------------------------------
// host launch
// ---------------------------------------------------------------------------
extern "C" void kernel_launch(void* const* d_in, const int* in_sizes, int n_in,
                              void* d_out, int out_size, void* d_ws, size_t ws_size,
                              hipStream_t stream) {
    const float* X       = (const float*)d_in[0];
    const float* E       = (const float*)d_in[1];
    const float* y       = (const float*)d_in[2];
    const float* nmask   = (const float*)d_in[3];
    const float* q_W     = (const float*)d_in[4];
    const float* q_b     = (const float*)d_in[5];
    const float* k_W     = (const float*)d_in[6];
    const float* k_b     = (const float*)d_in[7];
    const float* v_W     = (const float*)d_in[8];
    const float* v_b     = (const float*)d_in[9];
    const float* ea_add_W = (const float*)d_in[10];
    const float* ea_add_b = (const float*)d_in[11];
    const float* ea_mul_W = (const float*)d_in[12];
    const float* ea_mul_b = (const float*)d_in[13];
    const float* ge_add_W = (const float*)d_in[14];
    const float* ge_add_b = (const float*)d_in[15];
    const float* ge_mul_W = (const float*)d_in[16];
    const float* ge_mul_b = (const float*)d_in[17];
    const float* gn_add_W = (const float*)d_in[18];
    const float* gn_add_b = (const float*)d_in[19];
    const float* gn_mul_W = (const float*)d_in[20];
    const float* gn_mul_b = (const float*)d_in[21];
    const float* gs_W    = (const float*)d_in[22];
    const float* gs_b    = (const float*)d_in[23];
    const float* na_W    = (const float*)d_in[24];
    const float* na_b    = (const float*)d_in[25];
    const float* eag_W   = (const float*)d_in[26];
    const float* eag_b   = (const float*)d_in[27];
    const float* no_W    = (const float*)d_in[28];
    const float* no_b    = (const float*)d_in[29];
    const float* eo_W    = (const float*)d_in[30];
    const float* eo_b    = (const float*)d_in[31];
    const float* go1_W   = (const float*)d_in[32];
    const float* go1_b   = (const float*)d_in[33];
    const float* go2_W   = (const float*)d_in[34];
    const float* go2_b   = (const float*)d_in[35];
    (void)in_sizes; (void)n_in; (void)out_size; (void)ws_size;

    // output layout: updated_X | updated_E | uy
    float* outX  = (float*)d_out;                    // 131072
    float* outE  = outX + (size_t)BB * NN * DX;      // 16777216
    float* outUY = outE + (size_t)BB * NN * NN * DE; // 512

    // workspace layout (~1.8 MB)
    char* ws = (char*)d_ws;
    size_t off = 0;
    auto alloc = [&](size_t bytes) { void* p = ws + off; off += (bytes + 255) & ~(size_t)255; return p; };
    float* Q  = (float*)alloc(sizeof(float) * BB * NN * DX);
    float* Km = (float*)alloc(sizeof(float) * BB * NN * DX);
    float* V  = (float*)alloc(sizeof(float) * BB * NN * DX);
    float* geA = (float*)alloc(sizeof(float) * BB * DX);
    float* geM = (float*)alloc(sizeof(float) * BB * DX);
    float* gnA = (float*)alloc(sizeof(float) * BB * DX);
    float* gnM = (float*)alloc(sizeof(float) * BB * DX);
    float* node_stats = (float*)alloc(sizeof(float) * BB * 4 * DX);
    float* edge_stats = (float*)alloc(sizeof(float) * BB * 4 * DE);
    unsigned short* eamul_bf = (unsigned short*)alloc(sizeof(unsigned short) * DX * DE);
    unsigned short* eaadd_bf = (unsigned short*)alloc(sizeof(unsigned short) * DX * DE);
    unsigned short* eo_bf    = (unsigned short*)alloc(sizeof(unsigned short) * DE * DX);
    unsigned short* qw_bf    = (unsigned short*)alloc(sizeof(unsigned short) * DX * DX);
    unsigned short* kw_bf    = (unsigned short*)alloc(sizeof(unsigned short) * DX * DX);
    unsigned short* vw_bf    = (unsigned short*)alloc(sizeof(unsigned short) * DX * DX);

    // K0: bf16 weight conversion
    cvt_bf16_kernel<<<(DX * DE + 255) / 256, 256, 0, stream>>>(ea_mul_W, eamul_bf, DX * DE);
    cvt_bf16_kernel<<<(DX * DE + 255) / 256, 256, 0, stream>>>(ea_add_W, eaadd_bf, DX * DE);
    cvt_bf16_kernel<<<(DE * DX + 255) / 256, 256, 0, stream>>>(eo_W, eo_bf, DE * DX);
    cvt_bf16_kernel<<<(DX * DX + 255) / 256, 256, 0, stream>>>(q_W, qw_bf, DX * DX);
    cvt_bf16_kernel<<<(DX * DX + 255) / 256, 256, 0, stream>>>(k_W, kw_bf, DX * DX);
    cvt_bf16_kernel<<<(DX * DX + 255) / 256, 256, 0, stream>>>(v_W, vw_bf, DX * DX);

    // K1: y-conditioned affine params
    y_affine_kernel<<<BB, 128, 0, stream>>>(y, ge_add_W, ge_add_b, ge_mul_W, ge_mul_b,
                                            gn_add_W, gn_add_b, gn_mul_W, gn_mul_b,
                                            geA, geM, gnA, gnM);

    // K2: Q, K, V (WMMA)
    qkv_wmma_kernel<<<(BB * NN) / 16, 256, 0, stream>>>(X, qw_bf, q_b, nmask, Q);
    qkv_wmma_kernel<<<(BB * NN) / 16, 256, 0, stream>>>(X, kw_bf, k_b, nmask, Km);
    qkv_wmma_kernel<<<(BB * NN) / 16, 256, 0, stream>>>(X, vw_bf, v_b, nmask, V);

    // K3: fused edge pipeline (WMMA) -> updated_E, updated_X
    edge_attn_kernel<<<BB * NN, 256, 0, stream>>>(E, Q, Km, V, nmask,
                                                  eamul_bf, eaadd_bf, eo_bf,
                                                  ea_mul_b, ea_add_b, eo_b,
                                                  geA, geM, gnA, gnM,
                                                  no_W, no_b, outX, outE);

    // K4-K6: statistics + head MLP -> uy
    xstats_kernel<<<BB, 128, 0, stream>>>(X, node_stats);
    estats_kernel<<<BB * DE, 256, 0, stream>>>(E, edge_stats);
    head_mlp_kernel<<<BB, 128, 0, stream>>>(y, node_stats, edge_stats,
                                            gs_W, gs_b, na_W, na_b, eag_W, eag_b,
                                            go1_W, go1_b, go2_W, go2_b, outUY);
}
